// DAGEdgeActionGNNv2_62380105007801
// MI455X (gfx1250) — compile-verified, matrix operand order checked by reading, not run on hardware
//
// DAGEdgeActionGNNv2 for MI455X (gfx1250, CDNA5, wave32).
// Matrix work -> v_wmma_f32_16x16x32_bf16 (bf16 in, f32 acc).
// Weights are pre-converted once to bf16 [N][K] (K-contiguous) and streamed
// into LDS with the Tensor Data Mover (tensor_load_to_lds + s_wait_tensorcnt),
// activations are converted with hardware f32->bf16 cvt and packed dword LDS
// stores. ~93 GFLOP / ~1.5 GB traffic: edge-MLP dominates; bf16 WMMA keeps it
// bandwidth-bound at 23.3 TB/s instead of f32-VALU compute-bound.
#include <hip/hip_runtime.h>
#include <hip/hip_bf16.h>

typedef __attribute__((ext_vector_type(16))) __bf16        v16bf;
typedef __attribute__((ext_vector_type(8)))  float         v8f;
typedef __attribute__((ext_vector_type(8)))  unsigned int  v8u;
typedef __attribute__((ext_vector_type(4)))  unsigned int  v4u;
typedef __attribute__((ext_vector_type(8)))  int           v8i;
typedef __attribute__((ext_vector_type(4)))  int           v4i;

#define GNN_B   512
#define GNN_NN  64
#define GNN_N   32768
#define GNN_D   128
#define GNN_H   4
#define GNN_HD  32
#define GNN_L   2
#define GNN_E   262144

static __device__ __forceinline__ unsigned short f2bf(float f) {
  return __builtin_bit_cast(unsigned short, (__bf16)f);  // hw cvt, RNE
}
static __device__ __forceinline__ unsigned pack2bf(float a, float b) {
  return (unsigned)f2bf(a) | ((unsigned)f2bf(b) << 16);
}

static __device__ __forceinline__ v8f vzero8() {
  v8f z;
#pragma unroll
  for (int i = 0; i < 8; ++i) z[i] = 0.0f;
  return z;
}

// 16x32 bf16 fragment from LDS rows stored K-contiguous. Serves A ([m][k])
// and B ([n][k], i.e. K-major transposed). ISA 7.12.2: lanes 0-15 row=lane,
// K in {0..7,16..23}; lanes 16-31 row=lane-16, K in {8..15,24..31}.
static __device__ __forceinline__ v16bf load_frag(const unsigned short* lds,
                                                  int stride, int row0,
                                                  int kofs, int lane) {
  int r  = row0 + (lane & 15);
  int kh = ((lane >> 4) & 1) * 8;
  const unsigned short* p = lds + r * stride + kofs + kh;
  v8u u;
#pragma unroll
  for (int i = 0; i < 8; ++i) {
    int k = (i < 4) ? (2 * i) : (2 * i + 8);
    u[i] = *(const unsigned int*)(p + k);
  }
  return __builtin_bit_cast(v16bf, u);
}

static __device__ __forceinline__ v8f wmma_bf16(v16bf a, v16bf b, v8f c) {
  return __builtin_amdgcn_wmma_f32_16x16x32_bf16(false, a, false, b,
                                                 (short)0, c, false, false);
}

// TDM: async-load a 2-D bf16 tile [tileD1 rows x tileD0 elems] (row stride
// strideElems) from global into LDS at lds_off. D# per ISA ch.8 (group0:
// count/lds/global/type; group1: data_size=2B, dims, stride; groups2,3 zero).
// This toolchain's builtin takes 6 args (4 descriptor groups + spare group +
// cache policy).
static __device__ __forceinline__ void tdm_load_2d(unsigned lds_off,
                                                   const void* gaddr,
                                                   unsigned tileD0,
                                                   unsigned tileD1,
                                                   unsigned long long strideElems) {
  unsigned long long ga = (unsigned long long)gaddr;
  v4u g0;
  g0[0] = 1u;                                            // count=1 (user D#)
  g0[1] = lds_off;                                       // lds_addr
  g0[2] = (unsigned)(ga & 0xFFFFFFFFull);                // global_addr[31:0]
  g0[3] = (unsigned)((ga >> 32) & 0x01FFFFFFull) | 0x80000000u;  // [56:32]|type=2
  v8i g1;
  g1[0] = (int)(1u << 16);                               // data_size=1 (2B)
  g1[1] = (int)((tileD0 & 0xFFFFu) << 16);               // tensor_dim0 lo
  g1[2] = (int)(((tileD0 >> 16) & 0xFFFFu) | ((tileD1 & 0xFFFFu) << 16));
  g1[3] = (int)(((tileD1 >> 16) & 0xFFFFu) | ((tileD0 & 0xFFFFu) << 16));
  g1[4] = (int)(tileD1 & 0xFFFFu);                       // tile_dim1 (dim2=0)
  g1[5] = (int)(strideElems & 0xFFFFFFFFull);            // dim0_stride lo
  g1[6] = (int)((strideElems >> 32) & 0xFFFFull);        // dim0_stride hi
  g1[7] = 0;
  v4i g2; g2[0] = g2[1] = g2[2] = g2[3] = 0;
  v4i g3; g3[0] = g3[1] = g3[2] = g3[3] = 0;
  v8i g4;
#pragma unroll
  for (int i = 0; i < 8; ++i) g4[i] = 0;
  __builtin_amdgcn_tensor_load_to_lds(g0, g1, g2, g3, g4, 0);
}

// ---------------------------------------------------------------- utilities
__global__ void k_gather_embed(const int* __restrict__ feat,
                               const float* __restrict__ table,
                               float* __restrict__ ne) {
  int i = blockIdx.x, d = threadIdx.x;
  ne[i * GNN_D + d] = table[(size_t)feat[i] * GNN_D + d];
}

__global__ void k_edge_init(const float* __restrict__ e1,
                            float* __restrict__ ee) {
  int gid = blockIdx.x * 256 + threadIdx.x;
  ee[gid] = e1[gid & (GNN_D - 1)];
}

// W[K][N] f32 -> Wt[N][K] bf16 (one-time, tiny)
__global__ void k_w2bft(const float* __restrict__ W,
                        unsigned short* __restrict__ o, int K, int N) {
  int gid = blockIdx.x * 256 + threadIdx.x;
  if (gid >= K * N) return;
  int k = gid / N, n = gid - k * N;
  o[n * K + k] = f2bf(W[gid]);
}

__global__ void k_scatter(const float* __restrict__ ee,
                          const int* __restrict__ dst,
                          float* __restrict__ na) {
  int gid = blockIdx.x * 256 + threadIdx.x;
  int e = gid >> 5, q = gid & 31;
  int n = dst[e];
  const float* s = ee + e * GNN_D + q * 4;
  float* p = na + n * GNN_D + q * 4;
#pragma unroll
  for (int j = 0; j < 4; ++j) atomicAdd(p + j, s[j]);
}

__global__ void k_reduce_graph(const float* __restrict__ na,
                               const float* __restrict__ ne,
                               float* __restrict__ eg,
                               float* __restrict__ ng) {
  int b = blockIdx.x, d = threadIdx.x;
  float se = 0.f, sn = 0.f;
  for (int i = 0; i < GNN_NN; ++i) {
    int r = (b * GNN_NN + i) * GNN_D + d;
    se += na[r];
    sn += ne[r];
  }
  eg[b * GNN_D + d] = se;
  ng[b * GNN_D + d] = sn;
}

// ------------------------------------------------------- fused 2-stage MLP
// out (+)= relu(gather(row) @ W1 + b1) @ W2 + b2 ; weights bf16 [128][K].
// MODE 0: EDGE  [sA[i0[r]], sA[i1[r]], sC[r]]   (KIN=384)
// MODE 1: NODE  [sA[r], sB[r], sC[r>>6]]        (KIN=384)
// MODE 2: GRAPH [sA[r], sB[r], sC[r]]           (KIN=384)
// MODE 3: DENSE sA[r]                            (KIN=128)
template <int MODE>
__global__ __launch_bounds__(256) void k_fused_mlp(
    const float* __restrict__ sA, const float* __restrict__ sB,
    const float* __restrict__ sC, const int* __restrict__ i0,
    const int* __restrict__ i1, const unsigned short* __restrict__ W1t,
    const float* __restrict__ b1, const unsigned short* __restrict__ W2t,
    const float* __restrict__ b2, float* __restrict__ outp, int residual) {
  constexpr int KIN = (MODE == 3) ? 128 : 384;
  __shared__ __attribute__((aligned(16))) unsigned short a_t[64 * 36];
  __shared__ __attribute__((aligned(16))) unsigned short w_t[128 * 32];
  __shared__ __attribute__((aligned(16))) unsigned short h_t[64 * 132];

  const int tid = threadIdx.x, lane = tid & 31, w = tid >> 5;
  const int wm = w >> 1, wn = w & 1;
  const int rowBase = blockIdx.x * 64;
  const unsigned wlds = (unsigned)(size_t)(const void*)w_t;

  v8f acc[4];
#pragma unroll
  for (int t = 0; t < 4; ++t) acc[t] = vzero8();

  const int lr = tid & 63, kk0 = (tid >> 6) * 8;
  const int grow = rowBase + lr;

  // ---- stage 1: [64,KIN] @ W1[KIN,128]
  for (int kb = 0; kb < KIN; kb += 32) {
    const int kg = kb + kk0;
    const float* src;
    if constexpr (MODE == 0) {
      if (kg < 128)       src = sA + i0[grow] * GNN_D + kg;
      else if (kg < 256)  src = sA + i1[grow] * GNN_D + (kg - 128);
      else                src = sC + grow * GNN_D + (kg - 256);
    } else if constexpr (MODE == 1) {
      if (kg < 128)       src = sA + grow * GNN_D + kg;
      else if (kg < 256)  src = sB + grow * GNN_D + (kg - 128);
      else                src = sC + (grow >> 6) * GNN_D + (kg - 256);
    } else if constexpr (MODE == 2) {
      if (kg < 128)       src = sA + grow * GNN_D + kg;
      else if (kg < 256)  src = sB + grow * GNN_D + (kg - 128);
      else                src = sC + grow * GNN_D + (kg - 256);
    } else {
      src = sA + grow * GNN_D + kg;
    }
#pragma unroll
    for (int j = 0; j < 8; j += 2)
      *(unsigned*)(a_t + lr * 36 + kk0 + j) = pack2bf(src[j], src[j + 1]);
    if (w == 0) {  // DMA this K-step's 128x32 bf16 weight slab into LDS
      tdm_load_2d(wlds, W1t + kb, 32u, 128u, (unsigned long long)KIN);
      __builtin_amdgcn_s_wait_tensorcnt(0);
    }
    __syncthreads();
    v16bf af = load_frag(a_t, 36, wm * 16, 0, lane);
#pragma unroll
    for (int t = 0; t < 4; ++t) {
      v16bf bf_ = load_frag(w_t, 32, wn * 64 + t * 16, 0, lane);
      acc[t] = wmma_bf16(af, bf_, acc[t]);
    }
    __syncthreads();
  }

  // ---- bias + relu -> hidden (bf16) in LDS
  const int rloc = ((lane >> 4) & 1) * 8, cl = lane & 15;
#pragma unroll
  for (int t = 0; t < 4; ++t) {
    int col = wn * 64 + t * 16 + cl;
    float bb = b1[col];
#pragma unroll
    for (int r = 0; r < 8; ++r) {
      float v = fmaxf(acc[t][r] + bb, 0.0f);
      h_t[(wm * 16 + rloc + r) * 132 + col] = f2bf(v);
    }
    acc[t] = vzero8();
  }
  __syncthreads();

  // ---- stage 2: hidden[64,128] @ W2[128,128]
  for (int kb = 0; kb < 128; kb += 32) {
    if (w == 0) {
      tdm_load_2d(wlds, W2t + kb, 32u, 128u, 128ull);
      __builtin_amdgcn_s_wait_tensorcnt(0);
    }
    __syncthreads();
    v16bf af = load_frag(h_t, 132, wm * 16, kb, lane);
#pragma unroll
    for (int t = 0; t < 4; ++t) {
      v16bf bf_ = load_frag(w_t, 32, wn * 64 + t * 16, 0, lane);
      acc[t] = wmma_bf16(af, bf_, acc[t]);
    }
    __syncthreads();
  }

  // ---- epilogue
#pragma unroll
  for (int t = 0; t < 4; ++t) {
    int col = wn * 64 + t * 16 + cl;
    float bb = b2[col];
#pragma unroll
    for (int r = 0; r < 8; ++r) {
      int gr = rowBase + wm * 16 + rloc + r;
      float v = acc[t][r] + bb;
      float* p = outp + gr * GNN_D + col;
      if (residual) *p = *p + v; else *p = v;
    }
  }
}

// ------------------------------------------------ generic GEMM + bias
// C[64,128 tile] = A[M,K](lda,f32) @ Wt^T (Wt bf16 [Nout][K]) + bias
__global__ __launch_bounds__(256) void k_gemm_bias(
    const float* __restrict__ A, int lda,
    const unsigned short* __restrict__ Wt, const float* __restrict__ bias,
    float* __restrict__ C, int ldc, int K) {
  __shared__ __attribute__((aligned(16))) unsigned short a_t[64 * 36];
  __shared__ __attribute__((aligned(16))) unsigned short w_t[128 * 32];
  const int tid = threadIdx.x, lane = tid & 31, w = tid >> 5;
  const int wm = w >> 1, wn = w & 1;
  const int rowBase = blockIdx.x * 64, colBase = blockIdx.y * 128;
  const unsigned wlds = (unsigned)(size_t)(const void*)w_t;

  v8f acc[4];
#pragma unroll
  for (int t = 0; t < 4; ++t) acc[t] = vzero8();

  const int lr = tid & 63, kk0 = (tid >> 6) * 8;
  for (int kb = 0; kb < K; kb += 32) {
    const float* src = A + (rowBase + lr) * lda + kb + kk0;
#pragma unroll
    for (int j = 0; j < 8; j += 2)
      *(unsigned*)(a_t + lr * 36 + kk0 + j) = pack2bf(src[j], src[j + 1]);
    if (w == 0) {
      tdm_load_2d(wlds, Wt + (size_t)colBase * K + kb, 32u, 128u,
                  (unsigned long long)K);
      __builtin_amdgcn_s_wait_tensorcnt(0);
    }
    __syncthreads();
    v16bf af = load_frag(a_t, 36, wm * 16, 0, lane);
#pragma unroll
    for (int t = 0; t < 4; ++t) {
      v16bf bf_ = load_frag(w_t, 32, wn * 64 + t * 16, 0, lane);
      acc[t] = wmma_bf16(af, bf_, acc[t]);
    }
    __syncthreads();
  }
  const int rloc = ((lane >> 4) & 1) * 8, cl = lane & 15;
#pragma unroll
  for (int t = 0; t < 4; ++t) {
    int col = colBase + wn * 64 + t * 16 + cl;
    float bb = bias[col];
#pragma unroll
    for (int r = 0; r < 8; ++r) {
      int gr = rowBase + wm * 16 + rloc + r;
      C[gr * ldc + col] = acc[t][r] + bb;
    }
  }
}

// --------------------------------------------- attention (one graph,head)
__global__ __launch_bounds__(128) void k_attn(const float* __restrict__ Q,
                                              const float* __restrict__ Km,
                                              const float* __restrict__ V,
                                              float* __restrict__ ctx) {
  __shared__ __attribute__((aligned(16))) unsigned short qa[64 * 36];
  __shared__ __attribute__((aligned(16))) unsigned short kt[64 * 36];
  __shared__ __attribute__((aligned(16))) unsigned short vt[32 * 68];
  __shared__ __attribute__((aligned(16))) unsigned short pa[64 * 68];
  __shared__ float sm[64 * 65];
  const int b = blockIdx.x >> 2, h = blockIdx.x & 3;
  const int tid = threadIdx.x, lane = tid & 31, w = tid >> 5;
  const int base = (b * GNN_NN) * GNN_D + h * GNN_HD;

  {  // each thread owns 16 K-contiguous elems of one row -> packed stores
    int q = tid >> 1, d0 = (tid & 1) * 16;
    const float* qp = Q + base + q * GNN_D + d0;
    const float* kp = Km + base + q * GNN_D + d0;
    const float* vp = V + base + q * GNN_D + d0;
#pragma unroll
    for (int j = 0; j < 16; j += 2) {
      *(unsigned*)(qa + q * 36 + d0 + j) = pack2bf(qp[j], qp[j + 1]);
      *(unsigned*)(kt + q * 36 + d0 + j) = pack2bf(kp[j], kp[j + 1]);
    }
#pragma unroll
    for (int j = 0; j < 16; ++j) vt[(d0 + j) * 68 + q] = f2bf(vp[j]);
  }
  __syncthreads();

  v8f acc[4];
#pragma unroll
  for (int t = 0; t < 4; ++t) acc[t] = vzero8();
  v16bf af = load_frag(qa, 36, w * 16, 0, lane);
#pragma unroll
  for (int t = 0; t < 4; ++t) {
    v16bf bf_ = load_frag(kt, 36, t * 16, 0, lane);  // B[k][n] = K[n][k]
    acc[t] = wmma_bf16(af, bf_, acc[t]);
  }
  const int rloc = ((lane >> 4) & 1) * 8, cl = lane & 15;
  const float sc = 0.17677669529663687f;  // 1/sqrt(32)
#pragma unroll
  for (int t = 0; t < 4; ++t)
#pragma unroll
    for (int r = 0; r < 8; ++r)
      sm[(w * 16 + rloc + r) * 65 + t * 16 + cl] = acc[t][r] * sc;
  __syncthreads();

  if (tid < 64) {
    float m = -3.4e38f;
    for (int k = 0; k < 64; ++k) m = fmaxf(m, sm[tid * 65 + k]);
    float s = 0.f;
    for (int k = 0; k < 64; ++k) {
      float ev = expf(sm[tid * 65 + k] - m);
      s += ev;
      sm[tid * 65 + k] = ev;
    }
    float inv = 1.f / s;
    for (int k = 0; k < 64; k += 2)
      *(unsigned*)(pa + tid * 68 + k) =
          pack2bf(sm[tid * 65 + k] * inv, sm[tid * 65 + k + 1] * inv);
  }
  __syncthreads();

  v8f acc2[2];
#pragma unroll
  for (int t = 0; t < 2; ++t) acc2[t] = vzero8();
#pragma unroll
  for (int kb = 0; kb < 64; kb += 32) {
    v16bf af2 = load_frag(pa, 68, w * 16, kb, lane);
#pragma unroll
    for (int t = 0; t < 2; ++t) {
      v16bf bf_ = load_frag(vt, 68, t * 16, kb, lane);  // B[k][n] = V[k][n]
      acc2[t] = wmma_bf16(af2, bf_, acc2[t]);
    }
  }
#pragma unroll
  for (int t = 0; t < 2; ++t)
#pragma unroll
    for (int r = 0; r < 8; ++r)
      ctx[(b * GNN_NN + w * 16 + rloc + r) * GNN_D + h * GNN_HD + t * 16 + cl] =
          acc2[t][r];
}

// ----------------------------------- edge actions: senders @ receivers^T
__global__ __launch_bounds__(128) void k_edge_act(const float* __restrict__ S,
                                                  const float* __restrict__ R,
                                                  const float* __restrict__ tp,
                                                  float* __restrict__ out) {
  __shared__ __attribute__((aligned(16))) unsigned short sa[64 * 36];
  __shared__ __attribute__((aligned(16))) unsigned short rt[64 * 36];
  const int b = blockIdx.x;
  const int tid = threadIdx.x, lane = tid & 31, w = tid >> 5;

  v8f acc[4];
#pragma unroll
  for (int t = 0; t < 4; ++t) acc[t] = vzero8();

  for (int kb = 0; kb < GNN_D; kb += 32) {
    {  // 64 rows x 32 k per step; thread pair covers one row (16 elems each)
      int q = tid >> 1, d0 = (tid & 1) * 16;
      const float* sp = S + (b * GNN_NN + q) * GNN_D + kb + d0;
      const float* rp = R + (b * GNN_NN + q) * GNN_D + kb + d0;
#pragma unroll
      for (int j = 0; j < 16; j += 2) {
        *(unsigned*)(sa + q * 36 + d0 + j) = pack2bf(sp[j], sp[j + 1]);
        *(unsigned*)(rt + q * 36 + d0 + j) = pack2bf(rp[j], rp[j + 1]);
      }
    }
    __syncthreads();
    v16bf af = load_frag(sa, 36, w * 16, 0, lane);
#pragma unroll
    for (int t = 0; t < 4; ++t) {
      v16bf bf_ = load_frag(rt, 36, t * 16, 0, lane);
      acc[t] = wmma_bf16(af, bf_, acc[t]);
    }
    __syncthreads();
  }
  float t0 = tp[0];
  float sp2 = fmaxf(t0, 0.f) + log1pf(expf(-fabsf(t0)));  // softplus
  float inv = 1.f / sp2;
  const int rloc = ((lane >> 4) & 1) * 8, cl = lane & 15;
#pragma unroll
  for (int t = 0; t < 4; ++t)
#pragma unroll
    for (int r = 0; r < 8; ++r)
      out[b * (GNN_NN * GNN_NN) + (w * 16 + rloc + r) * GNN_NN + t * 16 + cl] =
          acc[t][r] * inv;
}

// ---------------------------------------------------- stop head + logits
__global__ __launch_bounds__(128) void k_stop(const float* __restrict__ g,
                                              const float* __restrict__ W1,
                                              const float* __restrict__ b1,
                                              const float* __restrict__ W2,
                                              const float* __restrict__ b2,
                                              float* __restrict__ at) {
  __shared__ float gl[128];
  __shared__ float red[128];
  const int b = blockIdx.x, d = threadIdx.x;
  gl[d] = g[b * GNN_D + d];
  __syncthreads();
  float acc = b1[d];
  for (int k = 0; k < GNN_D; ++k) acc += gl[k] * W1[k * GNN_D + d];
  acc = fmaxf(acc, 0.f);
  red[d] = acc * W2[d];
  __syncthreads();
  for (int s = 64; s > 0; s >>= 1) {
    if (d < s) red[d] += red[d + s];
    __syncthreads();
  }
  if (d == 0) {
    float stop = red[0] + b2[0];
    float l = log1pf(expf(-fabsf(stop)));
    at[b * 3 + 0] = -__builtin_inff();
    at[b * 3 + 1] = fminf(-stop, 0.f) - l;  // log_sigmoid(-stop)
    at[b * 3 + 2] = fminf(stop, 0.f) - l;   // log_sigmoid(stop)
  }
}

// ------------------------------------------------------------------ host
extern "C" void kernel_launch(void* const* d_in, const int* in_sizes, int n_in,
                              void* d_out, int out_size, void* d_ws,
                              size_t ws_size, hipStream_t stream) {
  (void)in_sizes; (void)n_in; (void)out_size; (void)ws_size;
  const int*   feat  = (const int*)d_in[0];
  const int*   eidx  = (const int*)d_in[1];
  const float* table = (const float*)d_in[3];
  const float* eemb  = (const float*)d_in[4];
  const float* eW1 = (const float*)d_in[5],  *eb1 = (const float*)d_in[6];
  const float* eW2 = (const float*)d_in[7],  *eb2 = (const float*)d_in[8];
  const float* nW1 = (const float*)d_in[9],  *nb1 = (const float*)d_in[10];
  const float* nW2 = (const float*)d_in[11], *nb2 = (const float*)d_in[12];
  const float* gW1 = (const float*)d_in[13], *gb1 = (const float*)d_in[14];
  const float* gW2 = (const float*)d_in[15], *gb2 = (const float*)d_in[16];
  const float* projW = (const float*)d_in[17], *projb = (const float*)d_in[18];
  const float* Wq = (const float*)d_in[19], *Wk = (const float*)d_in[20];
  const float* Wv = (const float*)d_in[21];
  const float* bq = (const float*)d_in[22], *bk = (const float*)d_in[23];
  const float* bv = (const float*)d_in[24];
  const float* outW = (const float*)d_in[25], *outb = (const float*)d_in[26];
  const float* sW1 = (const float*)d_in[27], *sb1 = (const float*)d_in[28];
  const float* sW2 = (const float*)d_in[29], *sb2 = (const float*)d_in[30];
  const float* rW1 = (const float*)d_in[31], *rb1 = (const float*)d_in[32];
  const float* rW2 = (const float*)d_in[33], *rb2 = (const float*)d_in[34];
  const float* stW1 = (const float*)d_in[35], *stb1 = (const float*)d_in[36];
  const float* stW2 = (const float*)d_in[37], *stb2 = (const float*)d_in[38];
  const float* temp  = (const float*)d_in[39];

  const int* srcIdx = eidx;
  const int* dstIdx = eidx + GNN_E;

  char* ws = (char*)d_ws;
  const size_t SZ_EDGE = (size_t)GNN_E * GNN_D * 4;   // 134 MB
  const size_t SZ_NODE = (size_t)GNN_N * GNN_D * 4;   // 16 MB
  const size_t SZ_G    = (size_t)GNN_B * GNN_D * 4;   // 256 KB
  float* edge_embs = (float*)(ws);
  float* node_embs = (float*)(ws + SZ_EDGE);
  float* node_aggr = (float*)(ws + SZ_EDGE + SZ_NODE);
  float* gbuf      = (float*)(ws + SZ_EDGE + 2 * SZ_NODE);
  float* egbuf     = (float*)(ws + SZ_EDGE + 2 * SZ_NODE + SZ_G);
  float* ngbuf     = (float*)(ws + SZ_EDGE + 2 * SZ_NODE + 2 * SZ_G);
  // bf16 transposed weights live after the f32 buffers (~1.1 MB total)
  unsigned short* wbase = (unsigned short*)(ws + SZ_EDGE + 2 * SZ_NODE + 3 * SZ_G);
  size_t wcur = 0;
  auto walloc = [&](size_t elems) {
    unsigned short* p = wbase + wcur;
    wcur += elems;
    return p;
  };
  const size_t S1 = 3 * GNN_D * GNN_D, S2 = GNN_D * GNN_D;  // 49152, 16384
  unsigned short *eW1t[GNN_L], *eW2t[GNN_L], *nW1t[GNN_L], *nW2t[GNN_L];
  unsigned short *gW1t[GNN_L], *gW2t[GNN_L];
  for (int l = 0; l < GNN_L; ++l) {
    eW1t[l] = walloc(S1); eW2t[l] = walloc(S2);
    nW1t[l] = walloc(S1); nW2t[l] = walloc(S2);
    gW1t[l] = walloc(S1); gW2t[l] = walloc(S2);
  }
  unsigned short* projWt = walloc(S1);   // [384][128]
  unsigned short* Wqt = walloc(S2);
  unsigned short* Wkt = walloc(S2);
  unsigned short* Wvt = walloc(S2);
  unsigned short* outWt = walloc(S2);
  unsigned short* sW1t = walloc(S2);
  unsigned short* sW2t = walloc(S2);
  unsigned short* rW1t = walloc(S2);
  unsigned short* rW2t = walloc(S2);

  // Aliases inside the edge_embs region once message passing is done:
  float* X    = (float*)(ws);                        // [N, 3D]   48 MB
  float* Qb   = (float*)(ws + 50331648);             // [N, D]
  float* Kb   = (float*)(ws + 67108864);             // [N, D]
  float* Vb   = (float*)(ws + 83886080);             // [N, D]
  float* ctx  = (float*)(ws);                        // [N, D] (over X head)
  float* abuf = (float*)(ws + 16777216);             // [N, D]
  float* sendb = Qb;                                 // (Q dead after attn)
  float* recvb = Kb;                                 // (K dead after attn)

  float* outF = (float*)d_out;                 // [B*3] at, then [B*NN*NN]
  float* outEdge = outF + GNN_B * 3;

  // one-time weight conversion (f32 [K][N] -> bf16 [N][K])
  auto conv = [&](const float* src, unsigned short* dst, int K, int N) {
    k_w2bft<<<(K * N + 255) / 256, 256, 0, stream>>>(src, dst, K, N);
  };
  for (int l = 0; l < GNN_L; ++l) {
    conv(eW1 + l * S1, eW1t[l], 384, 128); conv(eW2 + l * S2, eW2t[l], 128, 128);
    conv(nW1 + l * S1, nW1t[l], 384, 128); conv(nW2 + l * S2, nW2t[l], 128, 128);
    conv(gW1 + l * S1, gW1t[l], 384, 128); conv(gW2 + l * S2, gW2t[l], 128, 128);
  }
  conv(projW, projWt, 128, 384);
  conv(Wq, Wqt, 128, 128); conv(Wk, Wkt, 128, 128); conv(Wv, Wvt, 128, 128);
  conv(outW, outWt, 128, 128);
  conv(sW1, sW1t, 128, 128); conv(sW2, sW2t, 128, 128);
  conv(rW1, rW1t, 128, 128); conv(rW2, rW2t, 128, 128);

  (void)hipMemsetAsync(gbuf, 0, SZ_G, stream);
  k_gather_embed<<<GNN_N, GNN_D, 0, stream>>>(feat, table, node_embs);
  k_edge_init<<<(GNN_E * GNN_D) / 256, 256, 0, stream>>>(eemb, edge_embs);

  for (int l = 0; l < GNN_L; ++l) {
    const int ob = l * GNN_D;
    k_fused_mlp<0><<<GNN_E / 64, 256, 0, stream>>>(
        node_embs, nullptr, edge_embs, srcIdx, dstIdx,
        eW1t[l], eb1 + ob, eW2t[l], eb2 + ob, edge_embs, 1);
    (void)hipMemsetAsync(node_aggr, 0, SZ_NODE, stream);
    k_scatter<<<(GNN_E * 32) / 256, 256, 0, stream>>>(edge_embs, dstIdx,
                                                      node_aggr);
    k_fused_mlp<1><<<GNN_N / 64, 256, 0, stream>>>(
        node_embs, node_aggr, gbuf, nullptr, nullptr,
        nW1t[l], nb1 + ob, nW2t[l], nb2 + ob, node_embs, 1);
    k_reduce_graph<<<GNN_B, GNN_D, 0, stream>>>(node_aggr, node_embs, egbuf,
                                                ngbuf);
    k_fused_mlp<2><<<GNN_B / 64, 256, 0, stream>>>(
        ngbuf, egbuf, gbuf, nullptr, nullptr,
        gW1t[l], gb1 + ob, gW2t[l], gb2 + ob, gbuf, 1);
  }

  // projection -> X [N, 3D]
  k_gemm_bias<<<dim3(GNN_N / 64, 3), 256, 0, stream>>>(
      node_embs, GNN_D, projWt, projb, X, 3 * GNN_D, GNN_D);
  // q/k/v heads
  k_gemm_bias<<<dim3(GNN_N / 64, 1), 256, 0, stream>>>(
      X, 3 * GNN_D, Wqt, bq, Qb, GNN_D, GNN_D);
  k_gemm_bias<<<dim3(GNN_N / 64, 1), 256, 0, stream>>>(
      X + GNN_D, 3 * GNN_D, Wkt, bk, Kb, GNN_D, GNN_D);
  k_gemm_bias<<<dim3(GNN_N / 64, 1), 256, 0, stream>>>(
      X + 2 * GNN_D, 3 * GNN_D, Wvt, bv, Vb, GNN_D, GNN_D);

  k_attn<<<GNN_B * GNN_H, 128, 0, stream>>>(Qb, Kb, Vb, ctx);

  k_gemm_bias<<<dim3(GNN_N / 64, 1), 256, 0, stream>>>(
      ctx, GNN_D, outWt, outb, abuf, GNN_D, GNN_D);
  k_fused_mlp<3><<<GNN_N / 64, 256, 0, stream>>>(
      abuf, nullptr, nullptr, nullptr, nullptr, sW1t, sb1, sW2t, sb2, sendb, 0);
  k_fused_mlp<3><<<GNN_N / 64, 256, 0, stream>>>(
      abuf, nullptr, nullptr, nullptr, nullptr, rW1t, rb1, rW2t, rb2, recvb, 0);

  k_edge_act<<<GNN_B, 128, 0, stream>>>(sendb, recvb, temp, outEdge);
  k_stop<<<GNN_B, 128, 0, stream>>>(gbuf, stW1, stb1, stW2, stb2, outF);
}
